// CrossLayerAttention_29523605192739
// MI455X (gfx1250) — matrix operand order repaired and checked by reading, not compiled, not run on hardware
//
#include <hip/hip_runtime.h>
#include <hip/hip_bf16.h>

typedef _Float16 half8 __attribute__((ext_vector_type(8)));
typedef _Float16 v16h  __attribute__((ext_vector_type(16)));
typedef float    v8f   __attribute__((ext_vector_type(8)));

// ---------------------------------------------------------------- problem dims
#define KB   4
#define KT   2048
#define KD   1024
#define KH   16
#define KHD  64
#define KTOK (KB * KT)   // 8192 tokens

// ---------------------------------------------------------------- ws layout
static constexpr size_t SZ_X  = (size_t)KTOK * KD * sizeof(_Float16);  // 16 MB
static constexpr size_t SZ_W  = (size_t)KD * KD * sizeof(_Float16);    //  2 MB
static constexpr size_t OFF_XQ  = 0;
static constexpr size_t OFF_XKV = OFF_XQ  + SZ_X;
static constexpr size_t OFF_WQ  = OFF_XKV + SZ_X;
static constexpr size_t OFF_WK  = OFF_WQ  + SZ_W;
static constexpr size_t OFF_WV  = OFF_WK  + SZ_W;
static constexpr size_t OFF_WO  = OFF_WV  + SZ_W;
static constexpr size_t OFF_Q   = OFF_WO  + SZ_W;
static constexpr size_t OFF_K   = OFF_Q   + SZ_X;
static constexpr size_t OFF_VT  = OFF_K   + SZ_X;
static constexpr size_t OFF_O   = OFF_VT  + SZ_X;

// ---------------------------------------------------------------- helpers
// A-matrix fragment (16x32 f16, MxK): lane m = lane%16, half = lane/16.
// elements 0..7  hold K = half*8  + e   (16 contiguous bytes at p)
// elements 8..15 hold K = 16+half*8+e   (16 contiguous bytes at p+16)
__device__ __forceinline__ v16h ld_frag_a(const _Float16* p) {
  union { v16h v; half8 h[2]; } u;
  u.h[0] = *(const half8*)(p);
  u.h[1] = *(const half8*)(p + 16);
  return u.v;
}
// B-matrix fragment (32x16 f16, KxN): lane n = lane%16, half = lane/16.
// elements 0..7  hold K = half*8  + e   (16 contiguous bytes at p)
// elements 8..15 hold K = 16+half*8+e   (16 contiguous bytes at p+16)
__device__ __forceinline__ v16h ld_frag_b(const _Float16* p) {
  union { v16h v; half8 h[2]; } u;
  u.h[0] = *(const half8*)(p);
  u.h[1] = *(const half8*)(p + 16);
  return u.v;
}
__device__ __forceinline__ v8f wmma16(v16h a, v16h b, v8f c) {
  // v_wmma_f32_16x16x32_f16  (args: neg_a, A, neg_b, B, c_mod, C, reuse_a, reuse_b)
  return __builtin_amdgcn_wmma_f32_16x16x32_f16(false, a, false, b, (short)0, c,
                                                false, false);
}

// ---------------------------------------------------------------- fp32 -> fp16
__global__ __launch_bounds__(256) void cvt_f32_f16_kernel(
    const float* __restrict__ src, _Float16* __restrict__ dst, int n) {
  int i = blockIdx.x * blockDim.x + threadIdx.x;
  int stride = gridDim.x * blockDim.x;
  for (; i < n; i += stride) dst[i] = (_Float16)src[i];
}

// ---------------------------------------------------------------- projection GEMM
// Computes Y[tok, c] = sum_k X[tok,k] * W[c,k] + bias[c]
// WMMA orientation: A = W (M = channel, K = in-dim), B = X^T (K = in-dim, N = tok).
// Wave tile: 32 channels x 64 tokens (2x4 16x16 tiles). Block = 8 waves = 64 ch x 256 tok.
// MODE 0: f16 out, Y[tok*KD + c]
// MODE 1: f16 out, transposed Vt[(b*KD + c)*KT + t]   (tok = b*KT + t)
// MODE 2: f32 out, Y[tok*KD + c]
template <int MODE>
__global__ __launch_bounds__(256) void proj_wmma_kernel(
    const _Float16* __restrict__ X, const _Float16* __restrict__ W,
    const float* __restrict__ bias, void* __restrict__ Yv) {
  const int lane = threadIdx.x & 31;
  const int w    = threadIdx.x >> 5;
  const int wm   = w & 1;         // channel sub-block
  const int wn   = w >> 1;        // token sub-block
  const int cb   = blockIdx.x * 64 + wm * 32;    // wave channel base
  const int tb   = blockIdx.y * 256 + wn * 64;   // wave token base
  const int n16  = lane & 15;
  const int half = lane >> 4;

  v8f acc[2][4];
#pragma unroll
  for (int mt = 0; mt < 2; ++mt) {
    float bv[8];
#pragma unroll
    for (int r = 0; r < 8; ++r) bv[r] = bias[cb + mt * 16 + half * 8 + r];
#pragma unroll
    for (int nt = 0; nt < 4; ++nt)
#pragma unroll
      for (int r = 0; r < 8; ++r) acc[mt][nt][r] = bv[r];
  }

  for (int k0 = 0; k0 < KD; k0 += 32) {
    v16h af[2];
#pragma unroll
    for (int mt = 0; mt < 2; ++mt)
      af[mt] = ld_frag_a(W + (size_t)(cb + mt * 16 + n16) * KD + k0 + half * 8);
    if (k0 + 32 < KD)
      __builtin_prefetch((const void*)(X + (size_t)(tb + n16) * KD + k0 + 32), 0, 1);
#pragma unroll
    for (int nt = 0; nt < 4; ++nt) {
      v16h bf = ld_frag_b(X + (size_t)(tb + nt * 16 + n16) * KD + k0 + half * 8);
#pragma unroll
      for (int mt = 0; mt < 2; ++mt) acc[mt][nt] = wmma16(af[mt], bf, acc[mt][nt]);
    }
  }

  // D layout: M = half*8 + r (channel), N = lane%16 (token)
  if constexpr (MODE == 0) {
    _Float16* Y = (_Float16*)Yv;
#pragma unroll
    for (int mt = 0; mt < 2; ++mt)
#pragma unroll
      for (int nt = 0; nt < 4; ++nt) {
        const int tok = tb + nt * 16 + n16;
        half8 hv;
#pragma unroll
        for (int r = 0; r < 8; ++r) hv[r] = (_Float16)acc[mt][nt][r];
        *(half8*)(Y + (size_t)tok * KD + cb + mt * 16 + half * 8) = hv;
      }
  } else if constexpr (MODE == 1) {
    _Float16* Yt = (_Float16*)Yv;
#pragma unroll
    for (int mt = 0; mt < 2; ++mt)
#pragma unroll
      for (int nt = 0; nt < 4; ++nt) {
        const int tok = tb + nt * 16 + n16;
        const int bidx = tok >> 11;      // tok / KT
        const int t    = tok & (KT - 1); // tok % KT
#pragma unroll
        for (int r = 0; r < 8; ++r) {
          const int c = cb + mt * 16 + half * 8 + r;
          Yt[((size_t)bidx * KD + c) * KT + t] = (_Float16)acc[mt][nt][r];
        }
      }
  } else {
    float* Y = (float*)Yv;
#pragma unroll
    for (int mt = 0; mt < 2; ++mt)
#pragma unroll
      for (int nt = 0; nt < 4; ++nt) {
        const int tok = tb + nt * 16 + n16;
        float* p = Y + (size_t)tok * KD + cb + mt * 16 + half * 8;
        *(float4*)(p)     = make_float4(acc[mt][nt][0], acc[mt][nt][1],
                                        acc[mt][nt][2], acc[mt][nt][3]);
        *(float4*)(p + 4) = make_float4(acc[mt][nt][4], acc[mt][nt][5],
                                        acc[mt][nt][6], acc[mt][nt][7]);
      }
  }
}

// ---------------------------------------------------------------- banded flash attention
// One wave handles 32 query rows of one (b, h). Q,K layout [B*T, D] f16;
// V layout transposed Vt[(b*KD + h*64 + dim)*KT + t] f16. Online softmax in
// exp2 domain (scale = 1/sqrt(hd) * log2(e) folded into scores).
__global__ __launch_bounds__(256) void attn_wmma_kernel(
    const _Float16* __restrict__ Q, const _Float16* __restrict__ Km,
    const _Float16* __restrict__ Vt, _Float16* __restrict__ O,
    const int* __restrict__ la_ptr) {
  __shared__ _Float16 plds[8][32 * 32];  // per-wave 32x32 P staging (transpose C->A)

  const int la   = __builtin_amdgcn_readfirstlane(*la_ptr);  // uniform -> SGPR
  const int lane = threadIdx.x & 31;
  const int w    = threadIdx.x >> 5;
  const int gw   = blockIdx.x * 8 + w;
  const int qtiles = KT / 32;                 // 64
  const int qt = gw % qtiles;
  const int h  = (gw / qtiles) % KH;
  const int b  = gw / (qtiles * KH);
  const int qbase = qt * 32;
  const int n16  = lane & 15;
  const int half = lane >> 4;

  const _Float16* Qh = Q  + (size_t)b * KT * KD + h * KHD;
  const _Float16* Kh = Km + (size_t)b * KT * KD + h * KHD;
  const _Float16* Vh = Vt + ((size_t)b * KD + h * KHD) * KT;

  // Q fragments: A-role, rows = queries, K = head dim (64 = 2 slices of 32)
  v16h qf[2][2];
#pragma unroll
  for (int mt = 0; mt < 2; ++mt)
#pragma unroll
    for (int ks = 0; ks < 2; ++ks)
      qf[mt][ks] =
          ld_frag_a(Qh + (size_t)(qbase + mt * 16 + n16) * KD + ks * 32 + half * 8);

  float mrun[2][8], lrun[2][8];
  v8f oacc[2][4];
#pragma unroll
  for (int mt = 0; mt < 2; ++mt) {
#pragma unroll
    for (int r = 0; r < 8; ++r) { mrun[mt][r] = -3.0e38f; lrun[mt][r] = 0.0f; }
#pragma unroll
    for (int nto = 0; nto < 4; ++nto)
#pragma unroll
      for (int r = 0; r < 8; ++r) oacc[mt][nto][r] = 0.0f;
  }

  // 1/sqrt(64) * log2(e): softmax done with exp2 (v_exp_f32 is natively exp2)
  const float scale2 = 0.125f * 1.44269504088896340736f;
  int jend = qbase + 32 + la;
  if (jend > KT) jend = KT;

  for (int jb = 0; jb < jend; jb += 32) {
    // ---- S = Q K^T (2x2 tiles, K-dim = 64)
    v8f s[2][2];
#pragma unroll
    for (int mt = 0; mt < 2; ++mt)
#pragma unroll
      for (int nt = 0; nt < 2; ++nt)
#pragma unroll
        for (int r = 0; r < 8; ++r) s[mt][nt][r] = 0.0f;
#pragma unroll
    for (int ks = 0; ks < 2; ++ks)
#pragma unroll
      for (int nt = 0; nt < 2; ++nt) {
        v16h kf = ld_frag_b(Kh + (size_t)(jb + nt * 16 + n16) * KD + ks * 32 +
                            half * 8);
#pragma unroll
        for (int mt = 0; mt < 2; ++mt) s[mt][nt] = wmma16(qf[mt][ks], kf, s[mt][nt]);
      }

    // Tile fully inside the band iff max_j (= jb+31) <= min_i (= qbase) + la.
    const bool tmask = (jb + 31 > qbase + la);

    // ---- scale (+ mask only on the boundary tile) + online softmax
#pragma unroll
    for (int mt = 0; mt < 2; ++mt) {
      float rmax[8];
#pragma unroll
      for (int r = 0; r < 8; ++r) rmax[r] = -3.0e38f;
      if (tmask) {
#pragma unroll
        for (int nt = 0; nt < 2; ++nt) {
          const int j = jb + nt * 16 + n16;
#pragma unroll
          for (int r = 0; r < 8; ++r) {
            const int i = qbase + mt * 16 + half * 8 + r;
            float sv = s[mt][nt][r] * scale2;
            if (j > i + la) sv = -3.0e38f;
            s[mt][nt][r] = sv;
            rmax[r] = fmaxf(rmax[r], sv);
          }
        }
      } else {
#pragma unroll
        for (int nt = 0; nt < 2; ++nt)
#pragma unroll
          for (int r = 0; r < 8; ++r) {
            const float sv = s[mt][nt][r] * scale2;
            s[mt][nt][r] = sv;
            rmax[r] = fmaxf(rmax[r], sv);
          }
      }
      // row data lives in one 16-lane half: xor-reduce masks 8..1 stay in-half
#pragma unroll
      for (int r = 0; r < 8; ++r) {
#pragma unroll
        for (int off = 8; off >= 1; off >>= 1)
          rmax[r] = fmaxf(rmax[r], __shfl_xor(rmax[r], off, 32));
      }
#pragma unroll
      for (int r = 0; r < 8; ++r) {
        const float newm = fmaxf(mrun[mt][r], rmax[r]);
        const float corr = exp2f(mrun[mt][r] - newm);
        mrun[mt][r] = newm;
        lrun[mt][r] *= corr;
#pragma unroll
        for (int nto = 0; nto < 4; ++nto) oacc[mt][nto][r] *= corr;
        float rs = 0.0f;
#pragma unroll
        for (int nt = 0; nt < 2; ++nt) {
          const float sv = s[mt][nt][r];
          float p = exp2f(sv - newm);
          if (tmask && sv < -1.0e37f) p = 0.0f;  // fully-masked element
          plds[w][(mt * 16 + half * 8 + r) * 32 + nt * 16 + n16] = (_Float16)p;
          rs += p;
        }
#pragma unroll
        for (int off = 8; off >= 1; off >>= 1) rs += __shfl_xor(rs, off, 32);
        lrun[mt][r] += rs;
      }
    }

    // ---- P (A-role) from LDS, then O += P V   (LDS ops are in-order per wave)
    v16h pf[2];
#pragma unroll
    for (int mt = 0; mt < 2; ++mt)
      pf[mt] = ld_frag_a(&plds[w][(mt * 16 + n16) * 32 + half * 8]);
#pragma unroll
    for (int nto = 0; nto < 4; ++nto) {
      v16h vf = ld_frag_b(Vh + (size_t)(nto * 16 + n16) * KT + jb + half * 8);
#pragma unroll
      for (int mt = 0; mt < 2; ++mt) oacc[mt][nto] = wmma16(pf[mt], vf, oacc[mt][nto]);
    }
  }

  // ---- normalize (one reciprocal per row) + store O[tok*KD + h*64 + dim] (f16)
#pragma unroll
  for (int mt = 0; mt < 2; ++mt)
#pragma unroll
    for (int r = 0; r < 8; ++r) {
      const int row = qbase + mt * 16 + half * 8 + r;
      const float inv = 1.0f / fmaxf(lrun[mt][r], 1e-20f);
#pragma unroll
      for (int nto = 0; nto < 4; ++nto) {
        const float v = oacc[mt][nto][r] * inv;
        O[((size_t)(b * KT + row)) * KD + h * KHD + nto * 16 + n16] = (_Float16)v;
      }
    }
}

// ---------------------------------------------------------------- launch
extern "C" void kernel_launch(void* const* d_in, const int* in_sizes, int n_in,
                              void* d_out, int out_size, void* d_ws, size_t ws_size,
                              hipStream_t stream) {
  const float* query     = (const float*)d_in[0];
  const float* key_value = (const float*)d_in[1];
  const float* Wq = (const float*)d_in[2];
  const float* bq = (const float*)d_in[3];
  const float* Wk = (const float*)d_in[4];
  const float* bk = (const float*)d_in[5];
  const float* Wv = (const float*)d_in[6];
  const float* bv = (const float*)d_in[7];
  const float* Wo = (const float*)d_in[8];
  const float* bo = (const float*)d_in[9];
  const int*   la = (const int*)d_in[10];

  char* ws = (char*)d_ws;
  _Float16* Xq16  = (_Float16*)(ws + OFF_XQ);
  _Float16* Xkv16 = (_Float16*)(ws + OFF_XKV);
  _Float16* Wq16  = (_Float16*)(ws + OFF_WQ);
  _Float16* Wk16  = (_Float16*)(ws + OFF_WK);
  _Float16* Wv16  = (_Float16*)(ws + OFF_WV);
  _Float16* Wo16  = (_Float16*)(ws + OFF_WO);
  _Float16* Q16   = (_Float16*)(ws + OFF_Q);
  _Float16* K16   = (_Float16*)(ws + OFF_K);
  _Float16* Vt16  = (_Float16*)(ws + OFF_VT);
  _Float16* O16   = (_Float16*)(ws + OFF_O);

  const int nX = KTOK * KD;  // 8,388,608
  const int nW = KD * KD;    // 1,048,576
  cvt_f32_f16_kernel<<<1024, 256, 0, stream>>>(query,     Xq16,  nX);
  cvt_f32_f16_kernel<<<1024, 256, 0, stream>>>(key_value, Xkv16, nX);
  cvt_f32_f16_kernel<<<256,  256, 0, stream>>>(Wq, Wq16, nW);
  cvt_f32_f16_kernel<<<256,  256, 0, stream>>>(Wk, Wk16, nW);
  cvt_f32_f16_kernel<<<256,  256, 0, stream>>>(Wv, Wv16, nW);
  cvt_f32_f16_kernel<<<256,  256, 0, stream>>>(Wo, Wo16, nW);

  dim3 pgrid(KD / 64, KTOK / 256);  // (16, 32)
  proj_wmma_kernel<0><<<pgrid, 256, 0, stream>>>(Xq16,  Wq16, bq, (void*)Q16);
  proj_wmma_kernel<0><<<pgrid, 256, 0, stream>>>(Xkv16, Wk16, bk, (void*)K16);
  proj_wmma_kernel<1><<<pgrid, 256, 0, stream>>>(Xkv16, Wv16, bv, (void*)Vt16);

  const int nwaves = KB * KH * (KT / 32);  // 4096 waves, 8 per block
  attn_wmma_kernel<<<nwaves / 8, 256, 0, stream>>>(Q16, K16, Vt16, O16, la);

  proj_wmma_kernel<2><<<pgrid, 256, 0, stream>>>(O16, Wo16, bo, d_out);
}
